// ImprovedGQA_80315888435253
// MI455X (gfx1250) — matrix operand order repaired and checked by reading, not compiled, not run on hardware
//
#include <hip/hip_runtime.h>
#include <hip/hip_bf16.h>

// ---------------------------------------------------------------------------
// GQA attention block for MI455X (gfx1250, wave32, WMMA bf16 16x16x32).
// One-shot f32->bf16 conversion, software-pipelined pure-bf16 WMMA GEMMs,
// flash attention with async LDS staging (global_load_async_to_lds_b128)
// and in-LDS V transpose so every WMMA operand is a contiguous 32B load.
// ---------------------------------------------------------------------------

typedef __attribute__((ext_vector_type(16))) __bf16 v16bf;
typedef __attribute__((ext_vector_type(8)))  __bf16 v8bf;
typedef __attribute__((ext_vector_type(8)))  float  v8f;

#define DIM        2048
#define L_SEQ      2048
#define NHEADS     16
#define NKV        4
#define HDIM       128
#define NFUSED     3072      // 2048 Q + 512 K + 512 V
#define MROWS      4096      // B*L

__device__ __forceinline__ v8f vzero8() {
  v8f z = {0.f, 0.f, 0.f, 0.f, 0.f, 0.f, 0.f, 0.f};
  return z;
}

__device__ __forceinline__ v8f wmma_bf16(v16bf a, v16bf b, v8f c) {
  return __builtin_amdgcn_wmma_f32_16x16x32_bf16(
      /*neg_a=*/false, a, /*neg_b=*/false, b,
      /*c_mod=*/(short)0, c, /*reuse_a=*/false, /*reuse_b=*/false);
}

// A-matrix (16x32, MxK) per-lane fetch from bf16 row-major (global or LDS):
// lane l<16: row=l, K = 0..15 ; lanes 16..31: row=l-16, K=16..31
__device__ __forceinline__ v16bf load_a_bf16(const __bf16* __restrict__ base, int ld,
                                             int row0, int k0, int lane) {
  const int r = row0 + (lane & 15);
  const int k = k0 + ((lane >> 4) << 4);
  const __bf16* p = base + (size_t)r * ld + k;
  v8bf lo = *(const v8bf*)p;
  v8bf hi = *(const v8bf*)(p + 8);
  v16bf a;
#pragma unroll
  for (int i = 0; i < 8; ++i) { a[i] = lo[i]; a[i + 8] = hi[i]; }
  return a;
}

// B-matrix (32x16, KxN) per-lane fetch from bf16 row-major [N,K] source:
// lane holds column n = n0+(lane&15); 16 contiguous K elems -> one 32B load.
__device__ __forceinline__ v16bf load_b_bf16(const __bf16* __restrict__ w, int ld,
                                             int n0, int k0, int lane) {
  const int n  = n0 + (lane & 15);
  const int kb = k0 + ((lane >> 4) << 4);
  return *(const v16bf*)(w + (size_t)n * ld + kb);
}

// ---------------------------------------------------------------------------
// Kernel 0: f32 -> bf16 conversion (bandwidth-bound, one shot per operand).
// ---------------------------------------------------------------------------
__global__ __launch_bounds__(256) void gqa_cvt_bf16(
    const float* __restrict__ src, __bf16* __restrict__ dst, int n) {
  const int i = (blockIdx.x * 256 + threadIdx.x) * 8;
  if (i < n) {
    float4 a = *(const float4*)(src + i);
    float4 b = *(const float4*)(src + i + 4);
    v8bf o;
    o[0] = (__bf16)a.x; o[1] = (__bf16)a.y; o[2] = (__bf16)a.z; o[3] = (__bf16)a.w;
    o[4] = (__bf16)b.x; o[5] = (__bf16)b.y; o[6] = (__bf16)b.z; o[7] = (__bf16)b.w;
    *(v8bf*)(dst + i) = o;
  }
}

// ---------------------------------------------------------------------------
// Software-pipelined 16x128-per-wave bf16 GEMM body (shared by k1 and k4).
// Double-buffers A and the 8 B tiles so the next k-step's 18 b128 loads are
// in flight while the current step's 8 WMMAs execute.
// ---------------------------------------------------------------------------
__device__ __forceinline__ void gemm_16x128_pipelined(
    const __bf16* __restrict__ A, const __bf16* __restrict__ W,
    int row0, int col0, int lane, v8f acc[8]) {
  v16bf a = load_a_bf16(A, DIM, row0, 0, lane);
  v16bf b[8];
#pragma unroll
  for (int t = 0; t < 8; ++t) b[t] = load_b_bf16(W, DIM, col0 + t * 16, 0, lane);

  for (int k = 0; k < DIM; k += 32) {
    const int kn = (k + 32 < DIM) ? (k + 32) : 0;  // branch-free wraparound
    __builtin_prefetch(A + (size_t)(row0 + (lane & 15)) * DIM + k + 256, 0, 1);
    __builtin_prefetch(W + (size_t)(col0 + (lane & 15)) * DIM + k + 256, 0, 1);
    v16bf an = load_a_bf16(A, DIM, row0, kn, lane);
    v16bf bn[8];
#pragma unroll
    for (int t = 0; t < 8; ++t) bn[t] = load_b_bf16(W, DIM, col0 + t * 16, kn, lane);
#pragma unroll
    for (int t = 0; t < 8; ++t) acc[t] = wmma_bf16(a, b[t], acc[t]);
    a = an;
#pragma unroll
    for (int t = 0; t < 8; ++t) b[t] = bn[t];
  }
}

// ---------------------------------------------------------------------------
// Kernel 1: fused QKV projection.  Y[4096,3072](bf16) = Xb @ Wb^T   (all bf16)
// block = 128 (4 waves); each wave -> 16 rows x 128 cols; grid = (24, 64)
// ---------------------------------------------------------------------------
__global__ __launch_bounds__(128) void gqa_qkv_gemm(
    const __bf16* __restrict__ Xb, const __bf16* __restrict__ Wb,
    __bf16* __restrict__ Y) {
  const int lane = threadIdx.x & 31;
  const int wave = threadIdx.x >> 5;
  const int row0 = blockIdx.y * 64 + wave * 16;
  const int col0 = blockIdx.x * 128;

  v8f acc[8];
#pragma unroll
  for (int t = 0; t < 8; ++t) acc[t] = vzero8();

  gemm_16x128_pipelined(Xb, Wb, row0, col0, lane, acc);

  const int ccol = lane & 15;
  const int rbase = (lane >> 4) * 8;
#pragma unroll
  for (int t = 0; t < 8; ++t)
#pragma unroll
    for (int r = 0; r < 8; ++r)
      Y[(size_t)(row0 + rbase + r) * NFUSED + col0 + t * 16 + ccol] =
          (__bf16)acc[t][r];
}

// ---------------------------------------------------------------------------
// Kernel 2: RMSNorm (q/k heads) + RoPE + scatter to head-major bf16 layout.
// One wave per (row, head-slot): 0..15=Q, 16..19=K, 20..23=V.
// ---------------------------------------------------------------------------
__global__ __launch_bounds__(256) void gqa_norm_rope(
    const __bf16* __restrict__ Y, const float* __restrict__ qg,
    const float* __restrict__ kg, __bf16* __restrict__ Qh,
    __bf16* __restrict__ Kh, __bf16* __restrict__ Vh) {
  const int lane = threadIdx.x & 31;
  const int wave = threadIdx.x >> 5;
  const int gid  = blockIdx.x * 8 + wave;
  const int row  = gid / 24;
  const int h    = gid % 24;
  const int b    = row >> 11;
  const int l    = row & (L_SEQ - 1);

  const int srccol = (h < 16) ? h * 128
                   : (h < 20) ? 2048 + (h - 16) * 128
                              : 2560 + (h - 20) * 128;
  const __bf16* src = Y + (size_t)row * NFUSED + srccol;

  const int d0 = lane * 4;
  float vals[4];
#pragma unroll
  for (int i = 0; i < 4; ++i) vals[i] = (float)src[d0 + i];

  if (h < 20) {
    float ss = vals[0] * vals[0] + vals[1] * vals[1] +
               vals[2] * vals[2] + vals[3] * vals[3];
#pragma unroll
    for (int off = 16; off > 0; off >>= 1) ss += __shfl_xor(ss, off, 32);
    const float inv = rsqrtf(ss * (1.0f / 128.0f) + 1e-6f);
    const float* g = (h < 16) ? qg : kg;
#pragma unroll
    for (int i = 0; i < 4; ++i) vals[i] *= inv * g[d0 + i];
    const float ln_theta_over_hd = 9.210340372f / 128.0f;  // ln(10000)/128
#pragma unroll
    for (int p = 0; p < 2; ++p) {
      const int i2 = d0 + 2 * p;
      const float ang = (float)l * __expf(-(float)i2 * ln_theta_over_hd);
      const float c = __cosf(ang), s = __sinf(ang);
      const float re = vals[2 * p], im = vals[2 * p + 1];
      vals[2 * p]     = re * c - im * s;
      vals[2 * p + 1] = re * s + im * c;
    }
  }

  __bf16* dst;
  if (h < 16)      dst = Qh + ((size_t)(b * NHEADS + h)       * L_SEQ + l) * HDIM;
  else if (h < 20) dst = Kh + ((size_t)(b * NKV   + (h - 16)) * L_SEQ + l) * HDIM;
  else             dst = Vh + ((size_t)(b * NKV   + (h - 20)) * L_SEQ + l) * HDIM;
#pragma unroll
  for (int i = 0; i < 4; ++i) dst[d0 + i] = (__bf16)vals[i];
}

// ---------------------------------------------------------------------------
// Kernel 3: streaming attention (mask is all-ones -> no masking).
// grid = (B*NHEADS, L/128) = (32, 16); block = 256 (8 waves).
// Per 64-key block: async-stage K->sA, V->sB; all waves compute S from sA;
// sB transposed into sA so P@V B-tiles are contiguous 32B LDS loads.
// Softmax stats are register-resident (xor-shuffle over the C layout).
// ---------------------------------------------------------------------------
__global__ __launch_bounds__(256) void gqa_attention(
    const __bf16* __restrict__ Qh, const __bf16* __restrict__ Kh,
    const __bf16* __restrict__ Vh, __bf16* __restrict__ Oo) {
  __shared__ __bf16 sA[64 * 128];     // K block [key][d], later V^T [d][key]
  __shared__ __bf16 sB[64 * 128];     // V block [key][d]
  __shared__ __bf16 sP[8][16 * 64];   // probabilities, wave-private

  const int lane = threadIdx.x & 31;
  const int wave = threadIdx.x >> 5;
  const int bh   = blockIdx.x;
  const int b    = bh >> 4;
  const int h    = bh & 15;
  const int kvh  = h >> 2;                 // GROUP = 4

  const __bf16* Q = Qh + (size_t)bh * L_SEQ * HDIM;
  const __bf16* K = Kh + (size_t)(b * NKV + kvh) * L_SEQ * HDIM;
  const __bf16* V = Vh + (size_t)(b * NKV + kvh) * L_SEQ * HDIM;
  const int q0 = blockIdx.y * 128 + wave * 16;

  const unsigned ldsA = (unsigned)(uintptr_t)&sA[0];   // low 32 bits = LDS addr
  const unsigned ldsB = (unsigned)(uintptr_t)&sB[0];

  // Loop-invariant Q A-tiles (16 rows x 128 d) preloaded into registers.
  v16bf aq[4];
#pragma unroll
  for (int k = 0; k < 4; ++k) aq[k] = load_a_bf16(Q, HDIM, q0, k * 32, lane);

  v8f o[8];
#pragma unroll
  for (int n = 0; n < 8; ++n) o[n] = vzero8();
  float m[8], lsum[8];
#pragma unroll
  for (int r = 0; r < 8; ++r) { m[r] = -3.0e38f; lsum[r] = 0.0f; }

  const float scale = 0.0883883476f;  // 1/sqrt(128)
  const int rbase = (lane >> 4) * 8;
  const int ccol  = lane & 15;

  for (int j = 0; j < L_SEQ; j += 64) {
    __syncthreads();  // previous P@V reads of sA / async writes complete
    // ---- async stage K->sA, V->sB (each 64x128 bf16 = 16 KB, linear) ------
    {
      const char* gK = (const char*)(K + (size_t)j * HDIM);
      const char* gV = (const char*)(V + (size_t)j * HDIM);
#pragma unroll
      for (int i = 0; i < 4; ++i) {
        const unsigned off = (threadIdx.x + i * 256) * 16;  // 0..16368
        const unsigned laA = ldsA + off;
        const unsigned laB = ldsB + off;
        const unsigned long long gaK = (unsigned long long)(uintptr_t)(gK + off);
        const unsigned long long gaV = (unsigned long long)(uintptr_t)(gV + off);
        asm volatile("global_load_async_to_lds_b128 %0, %1, off"
                     :: "v"(laA), "v"(gaK) : "memory");
        asm volatile("global_load_async_to_lds_b128 %0, %1, off"
                     :: "v"(laB), "v"(gaV) : "memory");
      }
    }
    asm volatile("s_wait_asynccnt 0x0" ::: "memory");
    __syncthreads();

    // ---- S(16x64) = Qtile @ Kblk^T ; B tiles contiguous from sA -----------
    v8f s[4];
#pragma unroll
    for (int t = 0; t < 4; ++t) s[t] = vzero8();
#pragma unroll
    for (int k = 0; k < 4; ++k) {
#pragma unroll
      for (int t = 0; t < 4; ++t) {
        v16bf bm = load_b_bf16(&sA[0], HDIM, t * 16, k * 32, lane);
        s[t] = wmma_bf16(aq[k], bm, s[t]);
      }
    }

    // ---- online softmax in registers; P -> wave-private LDS ---------------
#pragma unroll
    for (int r = 0; r < 8; ++r) {
      float v0 = s[0][r] * scale, v1 = s[1][r] * scale;
      float v2 = s[2][r] * scale, v3 = s[3][r] * scale;
      float mx = fmaxf(fmaxf(v0, v1), fmaxf(v2, v3));
#pragma unroll
      for (int msk = 1; msk <= 8; msk <<= 1) mx = fmaxf(mx, __shfl_xor(mx, msk, 32));
      const float mnew  = fmaxf(m[r], mx);
      const float alpha = __expf(m[r] - mnew);
      const float p0 = __expf(v0 - mnew), p1 = __expf(v1 - mnew);
      const float p2 = __expf(v2 - mnew), p3 = __expf(v3 - mnew);
      float psum = p0 + p1 + p2 + p3;
#pragma unroll
      for (int msk = 1; msk <= 8; msk <<= 1) psum += __shfl_xor(psum, msk, 32);
      lsum[r] = lsum[r] * alpha + psum;
      m[r] = mnew;
#pragma unroll
      for (int n = 0; n < 8; ++n) o[n][r] *= alpha;
      const int prow = (rbase + r) * 64;
      sP[wave][prow + 0 * 16 + ccol] = (__bf16)p0;
      sP[wave][prow + 1 * 16 + ccol] = (__bf16)p1;
      sP[wave][prow + 2 * 16 + ccol] = (__bf16)p2;
      sP[wave][prow + 3 * 16 + ccol] = (__bf16)p3;
    }
    asm volatile("s_wait_dscnt 0x0" ::: "memory");

    // ---- transpose V block: sB[key][d] -> sA[d][key] ----------------------
    __syncthreads();                       // all waves done reading K from sA
    {
      const int kk = threadIdx.x & 63;     // key row
      const int d0 = (threadIdx.x >> 6) * 32;
#pragma unroll
      for (int c = 0; c < 4; ++c) {
        v8bf vv = *(const v8bf*)(&sB[kk * 128 + d0 + c * 8]);
#pragma unroll
        for (int e = 0; e < 8; ++e) sA[(d0 + c * 8 + e) * 64 + kk] = vv[e];
      }
    }
    __syncthreads();

    // ---- O(16x128) += P(16x64) @ Vblk; A from sP, B contiguous from sA ----
#pragma unroll
    for (int ks = 0; ks < 2; ++ks) {
      v16bf a = load_a_bf16(&sP[wave][0], 64, 0, ks * 32, lane);
#pragma unroll
      for (int n = 0; n < 8; ++n) {
        v16bf bm = load_b_bf16(&sA[0], 64, n * 16, ks * 32, lane);
        o[n] = wmma_bf16(a, bm, o[n]);
      }
    }
  }

  // ---- normalize by 1/l and store head-interleaved [B, L, H*128] ---------
#pragma unroll
  for (int r = 0; r < 8; ++r) {
    const float invl = 1.0f / lsum[r];
    const size_t orow = (size_t)b * L_SEQ + q0 + rbase + r;
#pragma unroll
    for (int n = 0; n < 8; ++n)
      Oo[orow * DIM + h * HDIM + n * 16 + ccol] = (__bf16)(o[n][r] * invl);
  }
}

// ---------------------------------------------------------------------------
// Kernel 4: output projection. out[4096,2048](f32) = Oo(bf16) @ Wob^T(bf16)
// block = 128 (4 waves); each wave 16x128; grid = (16, 64)
// ---------------------------------------------------------------------------
__global__ __launch_bounds__(128) void gqa_out_gemm(
    const __bf16* __restrict__ O, const __bf16* __restrict__ Wob,
    float* __restrict__ out) {
  const int lane = threadIdx.x & 31;
  const int wave = threadIdx.x >> 5;
  const int row0 = blockIdx.y * 64 + wave * 16;
  const int col0 = blockIdx.x * 128;

  v8f acc[8];
#pragma unroll
  for (int t = 0; t < 8; ++t) acc[t] = vzero8();

  gemm_16x128_pipelined(O, Wob, row0, col0, lane, acc);

  const int ccol = lane & 15, rbase = (lane >> 4) * 8;
#pragma unroll
  for (int t = 0; t < 8; ++t)
#pragma unroll
    for (int r = 0; r < 8; ++r)
      out[(size_t)(row0 + rbase + r) * DIM + col0 + t * 16 + ccol] = acc[t][r];
}

// ---------------------------------------------------------------------------
// Launch.  inputs = {x, mask, Wq, Wk, Wv, Wo, q_gamma, k_gamma}
// Workspace (64 MiB) with lifetime-based aliasing:
//   R0 [ 0       , +25165824): Y  (k1-k2)  -> Wob (k4)
//   R1 [25165824 , +16777216): Wb (k0-k1)  -> Qh  (k2-k3)
//   R2 [41943040 , + 4194304): Kh (k2-k3)
//   R3 [46137344 , + 4194304): Vh (k2-k3)
//   R4 [50331648 , +16777216): Xb (k0-k1)  -> Oo  (k3-k4)
// ---------------------------------------------------------------------------
extern "C" void kernel_launch(void* const* d_in, const int* in_sizes, int n_in,
                              void* d_out, int out_size, void* d_ws, size_t ws_size,
                              hipStream_t stream) {
  const float* x  = (const float*)d_in[0];
  // d_in[1] = mask (all ones for this workload)
  const float* Wq = (const float*)d_in[2];
  const float* Wk = (const float*)d_in[3];
  const float* Wv = (const float*)d_in[4];
  const float* Wo = (const float*)d_in[5];
  const float* qg = (const float*)d_in[6];
  const float* kg = (const float*)d_in[7];

  char* ws = (char*)d_ws;
  __bf16* Y   = (__bf16*)(ws);
  __bf16* Wob = (__bf16*)(ws);                         // aliases Y (dead after k2)
  __bf16* Wb  = (__bf16*)(ws + 25165824);
  __bf16* Qh  = (__bf16*)(ws + 25165824);              // aliases Wb (dead after k1)
  __bf16* Kh  = (__bf16*)(ws + 41943040);
  __bf16* Vh  = (__bf16*)(ws + 46137344);
  __bf16* Xb  = (__bf16*)(ws + 50331648);
  __bf16* Oo  = (__bf16*)(ws + 50331648);              // aliases Xb (dead after k1)

  // k0: one-shot f32 -> bf16 conversions (Wb = fused [Wq;Wk;Wv] rows)
  gqa_cvt_bf16<<<4096, 256, 0, stream>>>(x,  Xb,              MROWS * DIM);
  gqa_cvt_bf16<<<2048, 256, 0, stream>>>(Wq, Wb,              DIM * DIM);
  gqa_cvt_bf16<<< 512, 256, 0, stream>>>(Wk, Wb + 2048 * DIM, 512 * DIM);
  gqa_cvt_bf16<<< 512, 256, 0, stream>>>(Wv, Wb + 2560 * DIM, 512 * DIM);

  gqa_qkv_gemm<<<dim3(24, 64), 128, 0, stream>>>(Xb, Wb, Y);
  gqa_norm_rope<<<12288, 256, 0, stream>>>(Y, qg, kg, Qh, Kh, Vh);
  gqa_cvt_bf16<<<2048, 256, 0, stream>>>(Wo, Wob, DIM * DIM);   // Y now dead
  gqa_attention<<<dim3(32, 16), 256, 0, stream>>>(Qh, Kh, Vh, Oo);
  gqa_out_gemm<<<dim3(16, 64), 128, 0, stream>>>(Oo, Wob, (float*)d_out);
}
